// GIN_85736137163003
// MI455X (gfx1250) — matrix operand order repaired and checked by reading, not compiled, not run on hardware
//
#include <hip/hip_runtime.h>

#define NN 50000
#define DD 128
#define EE 800000
#define LL 3

typedef __attribute__((ext_vector_type(16))) __bf16 v16bf;
typedef __attribute__((ext_vector_type(8)))  __bf16 v8bf;
typedef __attribute__((ext_vector_type(8)))  float  v8f;

union V16U { v16bf v; v8bf h[2]; };

__device__ __forceinline__ v8f wmma_bf16(v16bf a, v16bf b, v8f c) {
  // D(f32 16x16) = A(16x32 bf16) * B(32x16 bf16) + C
  return __builtin_amdgcn_wmma_f32_16x16x32_bf16(false, a, false, b, (short)0, c,
                                                 false, false);
}

// ---------------- zero workspace ----------------
__global__ void k_zero(float* __restrict__ p, int n) {
  int i = blockIdx.x * blockDim.x + threadIdx.x;
  if (i < n) p[i] = 0.0f;
}

// ---------------- edge scatter: agg[dst] += h[src] ----------------
// one wave per edge; lane covers 4 contiguous floats; hw f32 atomics (L2-resident)
__global__ __launch_bounds__(256) void k_scatter(const float* __restrict__ h,
                                                 const int* __restrict__ ei,
                                                 float* __restrict__ agg) {
  int e = blockIdx.x * 8 + (threadIdx.x >> 5);
  if (e >= EE) return;
  int lane = threadIdx.x & 31;
  int src = ei[e];
  int dst = ei[EE + e];
  const float4 v = *(const float4*)(h + (size_t)src * DD + lane * 4);
  float* a = agg + (size_t)dst * DD + lane * 4;
  unsafeAtomicAdd(a + 0, v.x);
  unsafeAtomicAdd(a + 1, v.y);
  unsafeAtomicAdd(a + 2, v.z);
  unsafeAtomicAdd(a + 3, v.w);
}

// ---------------- one-time weight transpose+convert to bf16 ----------------
// Wt1[l][n<256][k<128] = W1[l][k][n];  Wt2[l][n<128][k<256] = W2[l][k][n]
__global__ void k_prep_weights(const float* __restrict__ W1, const float* __restrict__ W2,
                               __bf16* __restrict__ Wt1, __bf16* __restrict__ Wt2) {
  int idx = blockIdx.x * blockDim.x + threadIdx.x;
  const int per = DD * 2 * DD;  // 32768
  if (idx >= LL * per) return;
  int l = idx / per, rem = idx - l * per;
  {
    int n = rem >> 7, k = rem & 127;
    Wt1[idx] = (__bf16)W1[((size_t)l * DD + k) * (2 * DD) + n];
  }
  {
    int n = rem >> 8, k = rem & 255;
    Wt2[idx] = (__bf16)W2[((size_t)l * (2 * DD) + k) * DD + n];
  }
}

// ---------------- GEMM1: relu(bn(((1+eps)h + agg) @ W1 + b1)) -> z1 (bf16) ----
// block = 256 thr = 8 waves; block computes rows [16) x cols [256)
// A tile (16x128) built on the fly -> bf16 in padded LDS; each wave: 2 N-tiles
__global__ __launch_bounds__(256) void k_gemm1(
    const float* __restrict__ h, const float* __restrict__ agg,
    const __bf16* __restrict__ Wt1,
    const float* __restrict__ b1, const float* __restrict__ g1,
    const float* __restrict__ be1, const float* __restrict__ m1,
    const float* __restrict__ v1, const float* __restrict__ epsv,
    int layer, __bf16* __restrict__ z1) {
  __shared__ __bf16 As[16 * 136];  // +8 pad: conflict-free ds_read_b128
  const int tid = threadIdx.x;
  const int mbase = blockIdx.x * 16;
  const float ep = 1.0f + epsv[layer];
  {
    int base = tid * 8;          // 256 thr x 8 = 2048 = 16x128
    int row = base >> 7;
    int col = base & 127;
    const float* hp = h + (size_t)(mbase + row) * DD + col;
    const float* ap = agg + (size_t)(mbase + row) * DD + col;
    float4 h0 = *(const float4*)hp, h1 = *(const float4*)(hp + 4);
    float4 a0 = *(const float4*)ap, a1 = *(const float4*)(ap + 4);
    __bf16* d = As + row * 136 + col;
    d[0] = (__bf16)fmaf(ep, h0.x, a0.x); d[1] = (__bf16)fmaf(ep, h0.y, a0.y);
    d[2] = (__bf16)fmaf(ep, h0.z, a0.z); d[3] = (__bf16)fmaf(ep, h0.w, a0.w);
    d[4] = (__bf16)fmaf(ep, h1.x, a1.x); d[5] = (__bf16)fmaf(ep, h1.y, a1.y);
    d[6] = (__bf16)fmaf(ep, h1.z, a1.z); d[7] = (__bf16)fmaf(ep, h1.w, a1.w);
  }
  __syncthreads();
  const int wave = tid >> 5, lane = tid & 31;
  const int lhalf = lane >> 4, lmod = lane & 15;
  const int nb0 = wave * 32, nb1 = nb0 + 16;
  const __bf16* wbase = Wt1 + (size_t)layer * (2 * DD) * DD;
  const __bf16* w0 = wbase + (size_t)(nb0 + lmod) * DD;
  const __bf16* w1 = wbase + (size_t)(nb1 + lmod) * DD;
  v8f c0 = {};
  v8f c1 = {};
#pragma unroll
  for (int kt = 0; kt < 4; ++kt) {  // K = 128 in steps of 32
    const int kbA = kt * 32 + lhalf * 16;
    const __bf16* ar = As + lmod * 136 + kbA;
    V16U ua;
    ua.h[0] = *(const v8bf*)ar;          // K = kb..kb+7
    ua.h[1] = *(const v8bf*)(ar + 8);    // K = kb+8..kb+15
    const int kbB = kt * 32 + lhalf * 16;
    v16bf bb0 = *(const v16bf*)(w0 + kbB);  // 16 contiguous K (transposed W)
    v16bf bb1 = *(const v16bf*)(w1 + kbB);
    c0 = wmma_bf16(ua.v, bb0, c0);
    c1 = wmma_bf16(ua.v, bb1, c1);
  }
  const int lbase = layer * (2 * DD);
  const int rowb = mbase + lhalf * 8;
#pragma unroll
  for (int t = 0; t < 2; ++t) {
    const int col = (t ? nb1 : nb0) + lmod;
    const v8f& c = t ? c1 : c0;
    const float bb = b1[lbase + col];
    const float sc = rsqrtf(v1[lbase + col] + 1e-5f) * g1[lbase + col];
    const float mm = m1[lbase + col], bv = be1[lbase + col];
#pragma unroll
    for (int r = 0; r < 8; ++r) {
      float val = (c[r] + bb - mm) * sc + bv;
      val = fmaxf(val, 0.0f);
      z1[(size_t)(rowb + r) * (2 * DD) + col] = (__bf16)val;
    }
  }
}

// ---------------- GEMM2: z1 @ W2 + b2 (+BN+ReLU for layers 0,1) -> f32 h ----
// A tile staged via CDNA5 async global->LDS DMA (ASYNCcnt), no VGPR round-trip
__global__ __launch_bounds__(256) void k_gemm2(
    const __bf16* __restrict__ z1, const __bf16* __restrict__ Wt2,
    const float* __restrict__ b2, const float* __restrict__ go,
    const float* __restrict__ bo, const float* __restrict__ mo,
    const float* __restrict__ vo, int layer, int do_bn,
    float* __restrict__ out) {
  __shared__ __bf16 As[16 * 264];  // 16 rows x 256 (+8 pad)
  const int tid = threadIdx.x;
  const int mbase = blockIdx.x * 16;
  {
    int base = tid * 16;  // 256 thr x 16 = 4096 = 16x256
    int row = base >> 8;
    int col = base & 255;
    const __bf16* s = z1 + (size_t)(mbase + row) * (2 * DD) + col;
    __bf16* d = As + row * 264 + col;
    // LDS offset = low 32 bits of generic shared address (aperture rule)
    unsigned lds_off = (unsigned)(size_t)d;
    unsigned long long gaddr = (unsigned long long)(size_t)s;
    asm volatile("global_load_async_to_lds_b128 %0, %1, off"
                 :: "v"(lds_off), "v"(gaddr) : "memory");
    asm volatile("global_load_async_to_lds_b128 %0, %1, off offset:16"
                 :: "v"(lds_off), "v"(gaddr) : "memory");
    asm volatile("s_wait_asynccnt 0x0" ::: "memory");
  }
  __syncthreads();
  const int wave = tid >> 5, lane = tid & 31;
  const int lhalf = lane >> 4, lmod = lane & 15;
  const int nb = wave * 16;  // 8 waves x 16 = 128 cols
  const __bf16* w = Wt2 + (size_t)layer * DD * (2 * DD) + (size_t)(nb + lmod) * (2 * DD);
  v8f c = {};
#pragma unroll
  for (int kt = 0; kt < 8; ++kt) {  // K = 256 in steps of 32
    const int kbA = kt * 32 + lhalf * 16;
    const __bf16* ar = As + lmod * 264 + kbA;
    V16U ua;
    ua.h[0] = *(const v8bf*)ar;
    ua.h[1] = *(const v8bf*)(ar + 8);
    v16bf bb = *(const v16bf*)(w + kt * 32 + lhalf * 16);
    c = wmma_bf16(ua.v, bb, c);
  }
  const int col = nb + lmod;
  const int rowb = mbase + lhalf * 8;
  const float bb = b2[layer * DD + col];
  float sc = 1.0f, mm = 0.0f, bv = 0.0f;
  if (do_bn) {
    sc = rsqrtf(vo[layer * DD + col] + 1e-5f) * go[layer * DD + col];
    mm = mo[layer * DD + col];
    bv = bo[layer * DD + col];
  }
#pragma unroll
  for (int r = 0; r < 8; ++r) {
    float val = c[r] + bb;
    if (do_bn) {
      val = (val - mm) * sc + bv;
      val = fmaxf(val, 0.0f);
    }
    out[(size_t)(rowb + r) * DD + col] = val;
  }
}

// ---------------- log_softmax over 128-wide rows; one wave per row ----------
__global__ __launch_bounds__(256) void k_logsoftmax(const float* __restrict__ h,
                                                    float* __restrict__ out) {
  const int wave = threadIdx.x >> 5, lane = threadIdx.x & 31;
  const size_t row = (size_t)blockIdx.x * 8 + wave;
  const float4 v = *(const float4*)(h + row * DD + lane * 4);
  float mx = fmaxf(fmaxf(v.x, v.y), fmaxf(v.z, v.w));
#pragma unroll
  for (int off = 16; off >= 1; off >>= 1) mx = fmaxf(mx, __shfl_xor(mx, off, 32));
  float s = __expf(v.x - mx) + __expf(v.y - mx) + __expf(v.z - mx) + __expf(v.w - mx);
#pragma unroll
  for (int off = 16; off >= 1; off >>= 1) s += __shfl_xor(s, off, 32);
  const float lg = mx + __logf(s);
  float4 o;
  o.x = v.x - lg; o.y = v.y - lg; o.z = v.z - lg; o.w = v.w - lg;
  *(float4*)(out + row * DD + lane * 4) = o;
}

extern "C" void kernel_launch(void* const* d_in, const int* in_sizes, int n_in,
                              void* d_out, int out_size, void* d_ws, size_t ws_size,
                              hipStream_t stream) {
  const float* x   = (const float*)d_in[0];
  const int*   ei  = (const int*)d_in[1];
  const float* W1  = (const float*)d_in[2];
  const float* b1  = (const float*)d_in[3];
  const float* g1  = (const float*)d_in[4];
  const float* be1 = (const float*)d_in[5];
  const float* m1  = (const float*)d_in[6];
  const float* v1  = (const float*)d_in[7];
  const float* W2  = (const float*)d_in[8];
  const float* b2  = (const float*)d_in[9];
  const float* ep  = (const float*)d_in[10];
  const float* go  = (const float*)d_in[11];
  const float* bo  = (const float*)d_in[12];
  const float* mo  = (const float*)d_in[13];
  const float* vo  = (const float*)d_in[14];

  char* ws = (char*)d_ws;
  const size_t szH = (size_t)NN * DD * sizeof(float);        // 25.6 MB
  const size_t szZ = (size_t)NN * 2 * DD * sizeof(__bf16);   // 25.6 MB
  const size_t szW = (size_t)LL * DD * 2 * DD * sizeof(__bf16);
  float*  agg = (float*)ws;
  float*  h0  = (float*)(ws + szH);
  float*  h1  = (float*)(ws + 2 * szH);
  __bf16* z1  = (__bf16*)(ws + 3 * szH);
  __bf16* wt1 = (__bf16*)(ws + 3 * szH + szZ);
  __bf16* wt2 = (__bf16*)(ws + 3 * szH + szZ + szW);

  k_prep_weights<<<(LL * DD * 2 * DD + 255) / 256, 256, 0, stream>>>(W1, W2, wt1, wt2);

  const float* hcur = x;
  float* hb[2] = {h0, h1};
  for (int l = 0; l < LL; ++l) {
    k_zero<<<(NN * DD + 255) / 256, 256, 0, stream>>>(agg, NN * DD);
    k_scatter<<<EE / 8, 256, 0, stream>>>(hcur, ei, agg);
    k_gemm1<<<NN / 16, 256, 0, stream>>>(hcur, agg, wt1, b1, g1, be1, m1, v1, ep, l, z1);
    float* hn = hb[l & 1];
    k_gemm2<<<NN / 16, 256, 0, stream>>>(z1, wt2, b2, go, bo, mo, vo, l,
                                         (l < LL - 1) ? 1 : 0, hn);
    hcur = hn;
  }
  k_logsoftmax<<<NN / 8, 256, 0, stream>>>(hcur, (float*)d_out);
}
